// GMMPosteriorDist_86517821216271
// MI455X (gfx1250) — matrix operand order repaired and checked by reading, not compiled, not run on hardware
//
#include <hip/hip_runtime.h>
#include <math.h>

// ---------------------------------------------------------------------------
// Problem constants (from reference setup_inputs)
// ---------------------------------------------------------------------------
#define N_TRAJ 64
#define N_AUX  32
#define N_EMP  2048
#define DIM    256
#define NMC    100
#define M_TILES 7          // ceil(100/16)
#define N_TILES (N_EMP / 16)

typedef __attribute__((ext_vector_type(16))) __bf16 v16bf;
typedef __attribute__((ext_vector_type(8)))  float  v8f;

union BFrag { uint4 q[2]; v16bf v; };

// ---------------------------------------------------------------------------
// helpers
// ---------------------------------------------------------------------------
__device__ __forceinline__ unsigned bf16bits(float f) {
    union { float f; unsigned u; } in; in.f = f;
    unsigned u = in.u;
    u += 0x7FFFu + ((u >> 16) & 1u);          // round-to-nearest-even
    return u >> 16;
}
__device__ __forceinline__ __bf16 f32_to_bf16(float f) {
    unsigned short h = (unsigned short)bf16bits(f);
    union { unsigned short s; __bf16 b; } out; out.s = h;
    return out.b;
}

__device__ __forceinline__ unsigned long long splitmix64(unsigned long long x) {
    x += 0x9E3779B97F4A7C15ULL;
    x = (x ^ (x >> 30)) * 0xBF58476D1CE4E5B9ULL;
    x = (x ^ (x >> 27)) * 0x94D049BB133111EBULL;
    return x ^ (x >> 31);
}
__device__ __forceinline__ float u01(unsigned r) {
    return ((float)(r >> 8) + 0.5f) * (1.0f / 16777216.0f);   // (0,1)
}

// ---------------------------------------------------------------------------
// 1. scalar params
// ---------------------------------------------------------------------------
__global__ void params_kernel(const float* __restrict__ sigmas,
                              const int*   __restrict__ idx,
                              float* __restrict__ params) {
    if (threadIdx.x == 0) {
        int k = *idx;
        float s_km1 = 0.0f, s_k = 0.0f;
        for (int i = k; i < N_AUX; ++i)     s_km1 += sigmas[i];
        for (int i = k + 1; i < N_AUX; ++i) s_k   += sigmas[i];
        float ms  = sigmas[k] / s_km1;
        float var = sigmas[k] * s_k / s_km1;
        params[0] = ms;
        params[1] = var;
        params[2] = sqrtf(var);
        params[3] = -0.5f / var;
        params[4] = -0.5f * (float)DIM * logf(6.28318530718f * var);
        params[5] = ms * ms;
    }
}

// ---------------------------------------------------------------------------
// 2. b_k[t,d] = sum_a aux[t,a,d]
// ---------------------------------------------------------------------------
__global__ void bk_kernel(const float* __restrict__ aux, float* __restrict__ bk) {
    const int t = blockIdx.x, d = threadIdx.x;
    float s = 0.0f;
    for (int a = 0; a < N_AUX; ++a)
        s += aux[((size_t)t * N_AUX + a) * DIM + d];
    bk[t * DIM + d] = s;
}

// ---------------------------------------------------------------------------
// 3. logZ[t] = logsumexp_e hist[t,e]
// ---------------------------------------------------------------------------
__global__ void logz_kernel(const float* __restrict__ hist, float* __restrict__ logZ) {
    __shared__ float red[256];
    const int t = blockIdx.x, tid = threadIdx.x;
    float mx = -1e30f;
    for (int e = tid; e < N_EMP; e += 256) mx = fmaxf(mx, hist[t * N_EMP + e]);
    red[tid] = mx; __syncthreads();
    for (int s = 128; s > 0; s >>= 1) {
        if (tid < s) red[tid] = fmaxf(red[tid], red[tid + s]);
        __syncthreads();
    }
    mx = red[0]; __syncthreads();
    float sm = 0.0f;
    for (int e = tid; e < N_EMP; e += 256) sm += __expf(hist[t * N_EMP + e] - mx);
    red[tid] = sm; __syncthreads();
    for (int s = 128; s > 0; s >>= 1) {
        if (tid < s) red[tid] += red[tid + s];
        __syncthreads();
    }
    if (tid == 0) logZ[t] = mx + __logf(red[0]);
}

// ---------------------------------------------------------------------------
// 4. pack emp into bf16 WMMA B-fragment layout.
//    Fragment for (nt, lane, kt): e = nt*16 + (lane&15), d = kt*32 + (lane>>4)*16 + j
//    Stored so each lane's (kt) fragment is 32 contiguous bytes.
// ---------------------------------------------------------------------------
__global__ void pack_kernel(const float* __restrict__ emp, unsigned* __restrict__ epack) {
    const int nt  = blockIdx.x;        // 0..127
    const int tid = threadIdx.x;       // 256 = 8 kt * 32 lanes
    const int kt = tid >> 5, lane = tid & 31;
    const int e  = nt * 16 + (lane & 15);
    const int d0 = kt * 32 + ((lane >> 4) << 4);
    const float* src = emp + (size_t)e * DIM + d0;
    unsigned pk[8];
#pragma unroll
    for (int p = 0; p < 8; ++p)
        pk[p] = bf16bits(src[2 * p]) | (bf16bits(src[2 * p + 1]) << 16);
    uint4* dst = (uint4*)(epack + (((size_t)nt * 32 + lane) * 8 + kt) * 8);
    dst[0] = make_uint4(pk[0], pk[1], pk[2], pk[3]);
    dst[1] = make_uint4(pk[4], pk[5], pk[6], pk[7]);
}

// ---------------------------------------------------------------------------
// 5. m2[t,e] = ms^2 * sum_d (emp[e,d]-bk[t,d])^2
// ---------------------------------------------------------------------------
__global__ void m2_kernel(const float* __restrict__ emp,
                          const float* __restrict__ bk,
                          const float* __restrict__ params,
                          float* __restrict__ m2) {
    const int t = blockIdx.y;
    const int e = blockIdx.x * 128 + threadIdx.x;
    __shared__ float s_bk[DIM];
    for (int d = threadIdx.x; d < DIM; d += 128) s_bk[d] = bk[t * DIM + d];
    __syncthreads();
    float acc = 0.0f;
    const float* er = emp + (size_t)e * DIM;
    for (int d = 0; d < DIM; ++d) { float df = er[d] - s_bk[d]; acc = fmaf(df, df, acc); }
    m2[(size_t)t * N_EMP + e] = acc * params[5];
}

// ---------------------------------------------------------------------------
// 6. categorical sampling via Gumbel-max
// ---------------------------------------------------------------------------
__global__ void comp_kernel(const float* __restrict__ hist, int* __restrict__ comp) {
    const int t = blockIdx.x, m = blockIdx.y, tid = threadIdx.x;
    __shared__ float bval[128]; __shared__ int bidx[128];
    float best = -1e30f; int bi = 0;
    for (int e = tid; e < N_EMP; e += 128) {
        unsigned long long ctr = (((unsigned long long)(m * N_TRAJ + t)) << 32) | (unsigned)e;
        unsigned long long r = splitmix64(ctr ^ 0x0123456789ABCDEFULL);
        float g = -__logf(-__logf(u01((unsigned)r)));
        float v = hist[t * N_EMP + e] + g;
        if (v > best) { best = v; bi = e; }
    }
    bval[tid] = best; bidx[tid] = bi; __syncthreads();
    for (int s = 64; s > 0; s >>= 1) {
        if (tid < s && bval[tid + s] > bval[tid]) { bval[tid] = bval[tid + s]; bidx[tid] = bidx[tid + s]; }
        __syncthreads();
    }
    if (tid == 0) comp[m * N_TRAJ + t] = bidx[0];
}

// ---------------------------------------------------------------------------
// 7. samples + x2[m,t] = |s|^2 + sbk[m,t] = s . bk[t]
// ---------------------------------------------------------------------------
__global__ void samples_kernel(const float* __restrict__ emp,
                               const float* __restrict__ bk,
                               const int*   __restrict__ comp,
                               const float* __restrict__ params,
                               float* __restrict__ samples,
                               float* __restrict__ x2,
                               float* __restrict__ sbk) {
    const int t = blockIdx.x, m = blockIdx.y, d = threadIdx.x;
    __shared__ float red[DIM];
    const int ce = comp[m * N_TRAJ + t];
    const float bkv = bk[t * DIM + d];
    float mean = (emp[(size_t)ce * DIM + d] - bkv) * params[0];
    unsigned long long ctr = (((unsigned long long)(m * N_TRAJ + t)) << 32) | (unsigned)d;
    unsigned long long r = splitmix64(ctr ^ 0xFEDCBA9876543210ULL);
    float u1 = u01((unsigned)r), u2 = u01((unsigned)(r >> 32));
    float z = sqrtf(-2.0f * __logf(u1)) * __cosf(6.28318530718f * u2);
    float s = fmaf(params[2], z, mean);
    samples[((size_t)m * N_TRAJ + t) * DIM + d] = s;

    red[d] = s * s; __syncthreads();
    for (int st = 128; st > 0; st >>= 1) { if (d < st) red[d] += red[d + st]; __syncthreads(); }
    if (d == 0) x2[(size_t)m * N_TRAJ + t] = red[0];
    __syncthreads();
    red[d] = s * bkv; __syncthreads();
    for (int st = 128; st > 0; st >>= 1) { if (d < st) red[d] += red[d + st]; __syncthreads(); }
    if (d == 0) sbk[(size_t)m * N_TRAJ + t] = red[0];
}

// ---------------------------------------------------------------------------
// 8. Fused WMMA kernel.  Per (t, m-tile):
//      SE[m,e] = samples[m,t,:] . emp[e,:]         (bf16 WMMA, B pre-packed)
//      xm = ms*(SE - sbk[m,t])
//      log_q = lse_e( (hist-lZ) + nhlc + nhiv*m2[t,e] + nhiv*(x2b[m] - 2ms*SE) )
//    with x2b[m] = x2[m,t] + 2*ms*sbk[m,t].
// ---------------------------------------------------------------------------
__global__ __launch_bounds__(128) void logq_wmma_kernel(
    const float*    __restrict__ samples,
    const unsigned* __restrict__ epack,
    const float*    __restrict__ hist,
    const float*    __restrict__ qjh,
    const float*    __restrict__ logZ,
    const float*    __restrict__ x2,
    const float*    __restrict__ sbk,
    const float*    __restrict__ m2,
    const float*    __restrict__ params,
    float* __restrict__ out_logq,
    float* __restrict__ out_logqj) {

    const int t    = blockIdx.x / M_TILES;
    const int mt   = blockIdx.x % M_TILES;
    const int tid  = threadIdx.x;
    const int wave = tid >> 5;
    const int lane = tid & 31;
    const int half = lane >> 4;
    const int l16  = lane & 15;

    __shared__ float s_red[4][16][2];

    const float ms   = params[0];
    const float t2ms = 2.0f * ms;
    const float nhiv = params[3];
    const float nhlc = params[4];
    const float lZ   = logZ[t];

    // ---- A fragments: 16 rows of samples (bf16), contiguous k = 16*half + j layout (same as epack) ----
    const int   m_row = mt * 16 + l16;
    const int   m_cl  = m_row < NMC ? m_row : NMC - 1;
    const float amask = m_row < NMC ? 1.0f : 0.0f;
    const float* arow = samples + ((size_t)m_cl * N_TRAJ + t) * DIM;

    v16bf a[8];
#pragma unroll
    for (int kt = 0; kt < 8; ++kt) {
#pragma unroll
        for (int j = 0; j < 16; ++j) {
            const int krel = j + (half << 4);
            a[kt][j] = f32_to_bf16(amask * arow[kt * 32 + krel]);
        }
    }

    // per C-slot i: m = mt*16 + i + 8*half
    float x2b[8];
#pragma unroll
    for (int i = 0; i < 8; ++i) {
        int m = mt * 16 + i + 8 * half;
        int mc = m < NMC ? m : NMC - 1;
        x2b[i] = x2[(size_t)mc * N_TRAJ + t] + t2ms * sbk[(size_t)mc * N_TRAJ + t];
    }

    // lane-local online logsumexp state (branchless, no shuffles in the loop)
    float rmax[8], rsum[8];
#pragma unroll
    for (int i = 0; i < 8; ++i) { rmax[i] = -1e30f; rsum[i] = 0.0f; }

    for (int nt = wave; nt < N_TILES; nt += 4) {
        const int eB = nt * 16 + l16;
        const uint4* bb = (const uint4*)(epack + (((size_t)nt * 32 + lane) * 8) * 8);
        // unconditional prefetch of the next tile's fragments (wrapped, always valid)
        const int ntp = (nt + 4) & (N_TILES - 1);
        __builtin_prefetch(epack + (((size_t)ntp * 32 + lane) * 8) * 8, 0, 3);

        v8f c0 = {}, c1 = {};
#pragma unroll
        for (int kt = 0; kt < 8; kt += 2) {
            BFrag b0, b1;
            b0.q[0] = bb[2 * kt + 0]; b0.q[1] = bb[2 * kt + 1];
            b1.q[0] = bb[2 * kt + 2]; b1.q[1] = bb[2 * kt + 3];
            c0 = __builtin_amdgcn_wmma_f32_16x16x32_bf16(
                     false, a[kt],     false, b0.v, (short)0, c0, false, false);
            c1 = __builtin_amdgcn_wmma_f32_16x16x32_bf16(
                     false, a[kt + 1], false, b1.v, (short)0, c1, false, false);
        }

        const float base = (hist[t * N_EMP + eB] - lZ) + nhlc
                         + nhiv * m2[(size_t)t * N_EMP + eB];
#pragma unroll
        for (int i = 0; i < 8; ++i) {
            float v = base + nhiv * (x2b[i] - t2ms * (c0[i] + c1[i]));
            float nmax = fmaxf(rmax[i], v);
            rsum[i] = fmaf(rsum[i], __expf(rmax[i] - nmax), __expf(v - nmax));
            rmax[i] = nmax;
        }
    }

    // ---- reduce across the 16 lanes of each half (butterfly, once) ----
#pragma unroll
    for (int i = 0; i < 8; ++i) {
        float gm = rmax[i];
        for (int o = 1; o < 16; o <<= 1) gm = fmaxf(gm, __shfl_xor(gm, o, 32));
        float gs = rsum[i] * __expf(rmax[i] - gm);
        for (int o = 1; o < 16; o <<= 1) gs += __shfl_xor(gs, o, 32);
        rmax[i] = gm; rsum[i] = gs;
    }

    if (l16 == 0) {
#pragma unroll
        for (int i = 0; i < 8; ++i) {
            s_red[wave][i + 8 * half][0] = rmax[i];
            s_red[wave][i + 8 * half][1] = rsum[i];
        }
    }
    __syncthreads();
    if (wave == 0 && lane < 16) {
        const int m = mt * 16 + lane;
        if (m < NMC) {
            float gmax = -1e30f;
            for (int w = 0; w < 4; ++w) gmax = fmaxf(gmax, s_red[w][lane][0]);
            float gsum = 0.0f;
            for (int w = 0; w < 4; ++w) gsum += s_red[w][lane][1] * __expf(s_red[w][lane][0] - gmax);
            float lq = gmax + __logf(gsum);
            out_logq [(size_t)m * N_TRAJ + t] = lq;
            out_logqj[(size_t)m * N_TRAJ + t] = lq + qjh[t];
        }
    }
}

// ---------------------------------------------------------------------------
// launch
// ---------------------------------------------------------------------------
extern "C" void kernel_launch(void* const* d_in, const int* in_sizes, int n_in,
                              void* d_out, int out_size, void* d_ws, size_t ws_size,
                              hipStream_t stream) {
    const float* sigmas = (const float*)d_in[0];
    const float* aux    = (const float*)d_in[1];
    const float* hist   = (const float*)d_in[2];
    const float* qjh    = (const float*)d_in[3];
    const float* emp    = (const float*)d_in[4];
    const int*   idx    = (const int*)d_in[5];

    float* out     = (float*)d_out;
    float* samples = out;                                   // (100,64,256)
    float* logq    = out + (size_t)NMC * N_TRAJ * DIM;      // (100,64)
    float* logqj   = logq + (size_t)NMC * N_TRAJ;           // (100,64)

    float*    ws     = (float*)d_ws;
    float*    params = ws;                   // 16
    float*    logZ   = ws + 16;              // 64
    float*    bk     = ws + 80;              // 16384
    float*    x2     = ws + 16464;           // 6400
    float*    sbk    = ws + 22864;           // 6400
    float*    m2     = ws + 29264;           // 131072
    int*      comp   = (int*)(ws + 160336);  // 6400 ints
    unsigned* epack  = (unsigned*)(ws + 166736); // 262144 uints (1 MB)

    hipLaunchKernelGGL(params_kernel,  dim3(1),            dim3(32),  0, stream, sigmas, idx, params);
    hipLaunchKernelGGL(bk_kernel,      dim3(N_TRAJ),       dim3(DIM), 0, stream, aux, bk);
    hipLaunchKernelGGL(logz_kernel,    dim3(N_TRAJ),       dim3(256), 0, stream, hist, logZ);
    hipLaunchKernelGGL(pack_kernel,    dim3(N_TILES),      dim3(256), 0, stream, emp, epack);
    hipLaunchKernelGGL(comp_kernel,    dim3(N_TRAJ, NMC),  dim3(128), 0, stream, hist, comp);
    hipLaunchKernelGGL(m2_kernel,      dim3(16, N_TRAJ),   dim3(128), 0, stream, emp, bk, params, m2);
    hipLaunchKernelGGL(samples_kernel, dim3(N_TRAJ, NMC),  dim3(DIM), 0, stream,
                       emp, bk, comp, params, samples, x2, sbk);
    hipLaunchKernelGGL(logq_wmma_kernel, dim3(N_TRAJ * M_TILES), dim3(128), 0, stream,
                       samples, epack, hist, qjh, logZ, x2, sbk, m2, params, logq, logqj);
}
